// EncoderDecoder_44676249813333
// MI455X (gfx1250) — matrix operand order repaired and checked by reading, not compiled, not run on hardware
//
#include <hip/hip_runtime.h>
#include <hip/hip_bf16.h>
#include <math.h>

// ---------------- problem constants ----------------
constexpr int Bsz   = 32;
constexpr int SEQ   = 64;
constexpr int VOCAB = 512;
constexpr int EMB   = 256;
constexpr int HID   = 512;
constexpr int OSZ   = 512;
constexpr int VL    = 512;
constexpr int BETA  = 512;
constexpr int Lann  = 1024;           // (64/4)*(256/4) annotation vectors
constexpr int START = 141;
constexpr int ZLD   = EMB + OSZ + HID;   // 1280  : Z = [emb | O | h]  per batch row
constexpr int CLD   = HID + VL;          // 1024  : concat = [h | Ct]  per batch row

typedef __bf16 bf16;
typedef bf16  v16bf __attribute__((ext_vector_type(16)));
typedef bf16  bfx8  __attribute__((ext_vector_type(8)));
typedef float v8f   __attribute__((ext_vector_type(8)));

// ---------------- bf16 helpers (explicit RNE, no reliance on scalar __bf16 math) --
__device__ __forceinline__ bf16 f2bf(float f) {
  unsigned u = __builtin_bit_cast(unsigned, f);
  u += 0x7FFFu + ((u >> 16) & 1u);               // round-to-nearest-even
  unsigned short h = (unsigned short)(u >> 16);
  return __builtin_bit_cast(bf16, h);
}
__device__ __forceinline__ float bf2f(bf16 b) {
  unsigned short h = __builtin_bit_cast(unsigned short, b);
  unsigned u = ((unsigned)h) << 16;
  return __builtin_bit_cast(float, u);
}
__device__ __forceinline__ float sigm(float x) { return 1.0f / (1.0f + expf(-x)); }

// ---------------- WMMA fragment loaders (CDNA5 16x16x32 bf16 layouts) -------------
// A: row-major M x K.  Lane: m = lane&15, two 16B chunks at K=(lane>>4)*8 and +16.
__device__ __forceinline__ v16bf load_a_frag(const bf16* A, int lda, int m0, int kk, int lane) {
  int m    = m0 + (lane & 15);
  int koff = (lane >> 4) * 8;
  const bf16* p = A + (size_t)m * lda + kk + koff;
  bfx8 lo = *reinterpret_cast<const bfx8*>(p);
  bfx8 hi = *reinterpret_cast<const bfx8*>(p + 16);
  return __builtin_shufflevector(lo, hi, 0,1,2,3,4,5,6,7,8,9,10,11,12,13,14,15);
}
// B supplied transposed (BT: N x K row-major). Lane: n = lane&15, two 16B chunks
// at K=(lane>>4)*8 and +16 (same k order as A).
__device__ __forceinline__ v16bf load_b_frag(const bf16* BT, int ldb, int n0, int kk, int lane) {
  int n  = n0 + (lane & 15);
  int kb = (lane >> 4) * 8;
  const bf16* p = BT + (size_t)n * ldb + kk + kb;
  bfx8 lo = *reinterpret_cast<const bfx8*>(p);
  bfx8 hi = *reinterpret_cast<const bfx8*>(p + 16);
  return __builtin_shufflevector(lo, hi, 0,1,2,3,4,5,6,7,8,9,10,11,12,13,14,15);
}

// D tile store. f32 16x16 C/D layout: element (m = m0 + r + 8*(lane>>4), n = n0 + lane&15).
// MODE 0: fp32, transposed store DT[n*ldd + m] (+ optional bias[m])
// MODE 1: bf16, row-major store  D[m*ldd + n]
// MODE 2: bf16, tanh epilogue, transposed store DT[n*ldd + m]
template <int MODE>
__device__ __forceinline__ void store_tile(const v8f& c, int m0, int n0, void* D, int ldd,
                                           const float* bias, int lane) {
  int n  = n0 + (lane & 15);
  int mb = m0 + (lane >> 4) * 8;
  if (MODE == 0) {
    float* d = (float*)D;
#pragma unroll
    for (int r = 0; r < 8; ++r) {
      float v = c[r];
      int m = mb + r;
      if (bias) v += bias[m];
      d[(size_t)n * ldd + m] = v;
    }
  } else if (MODE == 1) {
    bf16* d = (bf16*)D;
#pragma unroll
    for (int r = 0; r < 8; ++r) d[(size_t)(mb + r) * ldd + n] = f2bf(c[r]);
  } else {
    bf16* d = (bf16*)D;
#pragma unroll
    for (int r = 0; r < 8; ++r) d[(size_t)n * ldd + (mb + r)] = f2bf(tanhf(c[r]));
  }
}

// ---------------- generic bf16 WMMA GEMM: D(MxN) = A(MxK) * B(KxN) ----------------
// One wave computes a 32x32 block (2x2 WMMA tiles, A-frag/B-frag reuse).
template <int MODE>
__global__ void wmma_gemm(const bf16* __restrict__ A, int lda,
                          const bf16* __restrict__ BT, int ldb,
                          void* __restrict__ D, int ldd, const float* __restrict__ bias,
                          int M, int N, int K) {
  int gtid = blockIdx.x * blockDim.x + threadIdx.x;
  int wid  = gtid >> 5;
  int lane = gtid & 31;
  int nG = N >> 5, mG = M >> 5;
  if (wid >= mG * nG) return;                 // wave-uniform exit; active waves EXEC=all-1s
  int m0 = (wid / nG) * 32;
  int n0 = (wid % nG) * 32;

  v8f c00 = {0.f,0.f,0.f,0.f,0.f,0.f,0.f,0.f};
  v8f c01 = c00, c10 = c00, c11 = c00;

  for (int kk = 0; kk < K; kk += 32) {
    v16bf a0 = load_a_frag(A, lda, m0,      kk, lane);
    v16bf a1 = load_a_frag(A, lda, m0 + 16, kk, lane);
    v16bf b0 = load_b_frag(BT, ldb, n0,      kk, lane);
    v16bf b1 = load_b_frag(BT, ldb, n0 + 16, kk, lane);
    c00 = __builtin_amdgcn_wmma_f32_16x16x32_bf16(false, a0, false, b0, (short)0, c00, false, false);
    c01 = __builtin_amdgcn_wmma_f32_16x16x32_bf16(false, a0, false, b1, (short)0, c01, false, false);
    c10 = __builtin_amdgcn_wmma_f32_16x16x32_bf16(false, a1, false, b0, (short)0, c10, false, false);
    c11 = __builtin_amdgcn_wmma_f32_16x16x32_bf16(false, a1, false, b1, (short)0, c11, false, false);
  }
  store_tile<MODE>(c00, m0,      n0,      D, ldd, bias, lane);
  store_tile<MODE>(c01, m0,      n0 + 16, D, ldd, bias, lane);
  store_tile<MODE>(c10, m0 + 16, n0,      D, ldd, bias, lane);
  store_tile<MODE>(c11, m0 + 16, n0 + 16, D, ldd, bias, lane);
}

// ---------------- encoder: stride-4 3x3 conv + relu -> Vbf[b][l][c] bf16 ----------
__global__ void conv_relu(const float* __restrict__ X, const float* __restrict__ Kw,
                          bf16* __restrict__ Vbf) {
  int idx = blockIdx.x * blockDim.x + threadIdx.x;
  if (idx >= Bsz * Lann * VL) return;
  int c = idx & (VL - 1);
  int l = (idx >> 9) & (Lann - 1);
  int b = idx >> 19;
  int oh = l >> 6, ow = l & 63;                 // Hc=16, Wc=64; SAME needs no padding here
  const float* xb = X + (size_t)b * (64 * 256) + (oh * 4) * 256 + ow * 4;
  const float* kc = Kw + c * 9;
  float acc = 0.f;
#pragma unroll
  for (int kh = 0; kh < 3; ++kh)
#pragma unroll
    for (int kw = 0; kw < 3; ++kw) acc += xb[kh * 256 + kw] * kc[kh * 3 + kw];
  Vbf[idx] = f2bf(fmaxf(acc, 0.f));
}

// ---------------- strided fp32 -> bf16 conversion (weight prep) -------------------
__global__ void cvt_strided(const float* __restrict__ src, int sld,
                            bf16* __restrict__ dst, int dld, int rows, int cols) {
  int idx = blockIdx.x * blockDim.x + threadIdx.x;
  if (idx >= rows * cols) return;
  int r = idx / cols, c = idx - r * cols;
  dst[(size_t)r * dld + c] = f2bf(src[(size_t)r * sld + c]);
}

// ---------------- decode-state init: Z=[E[:,141] | 0 | 0], c=0 --------------------
__global__ void init_state(const float* __restrict__ E, bf16* __restrict__ Zt,
                           float* __restrict__ cstate) {
  int idx = blockIdx.x * blockDim.x + threadIdx.x;
  if (idx < Bsz * ZLD) {
    int j = idx % ZLD;
    Zt[idx] = (j < EMB) ? f2bf(E[(size_t)j * VOCAB + START]) : f2bf(0.f);
  }
  if (idx < Bsz * HID) cstate[idx] = 0.f;
}

// ---------------- LSTM cell elementwise: c,h update; h -> Z.h and concat.h --------
__global__ void lstm_cell(const float* __restrict__ gT, float* __restrict__ cstate,
                          bf16* __restrict__ Zt, bf16* __restrict__ CC) {
  int idx = blockIdx.x * blockDim.x + threadIdx.x;
  if (idx >= Bsz * HID) return;
  int b = idx >> 9, hid = idx & (HID - 1);
  const float* g = gT + (size_t)b * (4 * HID);
  float gi = g[hid], gf = g[HID + hid], gg = g[2 * HID + hid], go = g[3 * HID + hid];
  float c = sigm(gf) * cstate[idx] + sigm(gi) * tanhf(gg);
  float h = sigm(go) * tanhf(c);
  cstate[idx] = c;
  bf16 hb = f2bf(h);
  Zt[(size_t)b * ZLD + EMB + OSZ + hid] = hb;
  CC[(size_t)b * CLD + hid] = hb;
}

// ---------------- attention scores: e[b][l] = sum_k tanh(VW+hW)*wa ----------------
__global__ void attn_score(const bf16* __restrict__ VWbf, const float* __restrict__ hW,
                           const float* __restrict__ wa, float* __restrict__ escore) {
  int gtid = blockIdx.x * blockDim.x + threadIdx.x;
  int w = gtid >> 5, lane = gtid & 31;
  if (w >= Bsz * Lann) return;
  int b = w >> 10;
  const bf16*  row = VWbf + (size_t)w * BETA;
  const float* hwb = hW + (size_t)b * BETA;
  float acc = 0.f;
  for (int k = lane; k < BETA; k += 32) acc += tanhf(bf2f(row[k]) + hwb[k]) * wa[k];
#pragma unroll
  for (int off = 16; off; off >>= 1) acc += __shfl_xor(acc, off, 32);
  if (lane == 0) escore[w] = acc;
}

// ---------------- row softmax over L=1024 (one block per batch row) ---------------
__global__ void softmax_rows(const float* __restrict__ e, float* __restrict__ alpha) {
  __shared__ float red[256];
  int b = blockIdx.x, tid = threadIdx.x;
  const float* eb = e + (size_t)b * Lann;
  float* ab = alpha + (size_t)b * Lann;
  float mx = -3.0e38f;
  for (int l = tid; l < Lann; l += 256) mx = fmaxf(mx, eb[l]);
  red[tid] = mx; __syncthreads();
  for (int s = 128; s; s >>= 1) { if (tid < s) red[tid] = fmaxf(red[tid], red[tid + s]); __syncthreads(); }
  mx = red[0]; __syncthreads();
  float sum = 0.f;
  for (int l = tid; l < Lann; l += 256) { float v = expf(eb[l] - mx); ab[l] = v; sum += v; }
  red[tid] = sum; __syncthreads();
  for (int s = 128; s; s >>= 1) { if (tid < s) red[tid] += red[tid + s]; __syncthreads(); }
  float inv = 1.0f / red[0];
  for (int l = tid; l < Lann; l += 256) ab[l] *= inv;
}

// ---------------- context: Ct[b][v] = sum_l alpha[b][l] * V[b][l][v] -> concat.Ct --
__global__ void ct_kernel(const float* __restrict__ alpha, const bf16* __restrict__ Vbf,
                          bf16* __restrict__ CC) {
  int idx = blockIdx.x * blockDim.x + threadIdx.x;
  if (idx >= Bsz * VL) return;
  int b = idx >> 9, v = idx & (VL - 1);
  const float* ab = alpha + (size_t)b * Lann;
  const bf16*  Vb = Vbf + (size_t)b * Lann * VL + v;
  float acc = 0.f;
  for (int l = 0; l < Lann; ++l) acc += ab[l] * bf2f(Vb[(size_t)l * VL]);
  CC[(size_t)b * CLD + HID + v] = f2bf(acc);
}

// ---------------- greedy feedback (faithful to Y_onehot[idx,:]=1 bug) -------------
// argmax per batch row; flags[v]=1 for any argmax'd vocab id; every batch row gets
// emb = sum_{flagged v} E[:,v] written into Z.emb.
__global__ void next_token(const float* __restrict__ logitsT, const float* __restrict__ E,
                           bf16* __restrict__ Zt) {
  __shared__ int flags[VOCAB];
  int tid = threadIdx.x;
  if (tid < VOCAB) flags[tid] = 0;
  __syncthreads();
  int b = tid >> 5, lane = tid & 31;
  const float* lb = logitsT + (size_t)b * (SEQ * VOCAB);   // base pre-offset by t*VOCAB
  float best = -3.0e38f; int bidx = 0;
#pragma unroll
  for (int i = 0; i < VOCAB / 32; ++i) {
    int v = lane + i * 32;
    float f = lb[v];
    if (f > best) { best = f; bidx = v; }
  }
#pragma unroll
  for (int off = 16; off; off >>= 1) {
    float ov = __shfl_xor(best, off, 32);
    int   oi = __shfl_xor(bidx, off, 32);
    if (ov > best || (ov == best && oi < bidx)) { best = ov; bidx = oi; }
  }
  if (lane == 0) flags[bidx] = 1;
  __syncthreads();
  if (tid < EMB) {
    float acc = 0.f;
    for (int v = 0; v < VOCAB; ++v)
      if (flags[v]) acc += E[(size_t)tid * VOCAB + v];
    bf16 hv = f2bf(acc);
    for (int bb = 0; bb < Bsz; ++bb) Zt[(size_t)bb * ZLD + tid] = hv;
  }
}

// =============================== host launch ======================================
static inline int gemm_grid(int M, int N) {          // blockDim = 128 (4 waves)
  int waves = (M / 32) * (N / 32);
  return (waves + 3) / 4;
}

extern "C" void kernel_launch(void* const* d_in, const int* in_sizes, int n_in,
                              void* d_out, int out_size, void* d_ws, size_t ws_size,
                              hipStream_t stream) {
  (void)in_sizes; (void)n_in; (void)out_size; (void)ws_size;
  const float* X       = (const float*)d_in[0];
  const float* conv_k  = (const float*)d_in[1];
  const float* E       = (const float*)d_in[2];
  const float* Wv      = (const float*)d_in[3];
  const float* Wh_att  = (const float*)d_in[4];
  const float* wa      = (const float*)d_in[5];
  const float* lstm_Wx = (const float*)d_in[6];
  const float* lstm_Wh = (const float*)d_in[7];
  const float* lstm_b  = (const float*)d_in[8];
  const float* W_O     = (const float*)d_in[9];
  const float* W_out   = (const float*)d_in[10];
  float* out = (float*)d_out;

  // ---- workspace carve (all 256B aligned) ----
  size_t off = 0;
  auto carve = [&](size_t bytes) -> char* {
    char* p = (char*)d_ws + off;
    off += (bytes + 255) & ~(size_t)255;
    return p;
  };
  bf16*  Vbf    = (bf16*) carve((size_t)Bsz * Lann * VL * 2);        // 32 MB
  bf16*  VWbf   = (bf16*) carve((size_t)Bsz * Lann * BETA * 2);      // 32 MB
  bf16*  Wstack = (bf16*) carve((size_t)4 * HID * ZLD * 2);          // [Wx|Wh] stacked? cols: EMB+OSZ+HID
  bf16*  WhA    = (bf16*) carve((size_t)BETA * HID * 2);
  bf16*  WOb    = (bf16*) carve((size_t)OSZ * CLD * 2);
  bf16*  Woutb  = (bf16*) carve((size_t)VOCAB * OSZ * 2);
  bf16*  Wvb    = (bf16*) carve((size_t)BETA * VL * 2);
  bf16*  Zt     = (bf16*) carve((size_t)Bsz * ZLD * 2);              // [emb|O|h] bf16 state
  bf16*  CC     = (bf16*) carve((size_t)Bsz * CLD * 2);              // [h|Ct]
  float* gT     = (float*)carve((size_t)Bsz * 4 * HID * 4);
  float* cstate = (float*)carve((size_t)Bsz * HID * 4);
  float* hW     = (float*)carve((size_t)Bsz * BETA * 4);
  float* escore = (float*)carve((size_t)Bsz * Lann * 4);
  float* alpha  = (float*)carve((size_t)Bsz * Lann * 4);

  // ---- encoder + weight prep (once) ----
  {
    int n = Bsz * Lann * VL;
    conv_relu<<<(n + 255) / 256, 256, 0, stream>>>(X, conv_k, Vbf);
  }
  auto cvt = [&](const float* s, int sld, bf16* d, int dld, int r, int c) {
    int n = r * c;
    cvt_strided<<<(n + 255) / 256, 256, 0, stream>>>(s, sld, d, dld, r, c);
  };
  cvt(lstm_Wx, EMB + OSZ, Wstack,             ZLD, 4 * HID, EMB + OSZ); // cols [0,768)
  cvt(lstm_Wh, HID,       Wstack + EMB + OSZ, ZLD, 4 * HID, HID);       // cols [768,1280)
  cvt(Wh_att,  HID,  WhA,   HID,  BETA,  HID);
  cvt(W_O,     CLD,  WOb,   CLD,  OSZ,   CLD);
  cvt(W_out,   OSZ,  Woutb, OSZ,  VOCAB, OSZ);
  cvt(Wv,      VL,   Wvb,   VL,   BETA,  VL);
  {
    int n = Bsz * ZLD;  // covers cstate range too
    init_state<<<(n + 255) / 256, 256, 0, stream>>>(E, Zt, cstate);
  }

  // ---- VW = V @ Wv^T : (32768 x 512) = (32768 x 512)x(512 x 512), bf16 out ----
  wmma_gemm<1><<<gemm_grid(Bsz * Lann, BETA), 128, 0, stream>>>(
      Vbf, VL, Wvb, VL, (void*)VWbf, BETA, nullptr, Bsz * Lann, BETA, VL);

  // ---- sequential decode: 64 steps ----
  for (int t = 0; t < SEQ; ++t) {
    // gates: gT[b][0:2048] = [Wx|Wh](2048x1280) @ Z^T + b
    wmma_gemm<0><<<gemm_grid(4 * HID, Bsz), 128, 0, stream>>>(
        Wstack, ZLD, Zt, ZLD, (void*)gT, 4 * HID, lstm_b, 4 * HID, Bsz, ZLD);
    // LSTM elementwise -> c, h (h into Z.h and concat.h as bf16)
    lstm_cell<<<(Bsz * HID + 255) / 256, 256, 0, stream>>>(gT, cstate, Zt, CC);
    // hW[b][k] = (h^T @ Wh_att^T): A = Wh_att, BT = concat.h rows
    wmma_gemm<0><<<gemm_grid(BETA, Bsz), 128, 0, stream>>>(
        WhA, HID, CC, CLD, (void*)hW, BETA, nullptr, BETA, Bsz, HID);
    // e[b][l] = sum_k tanh(VW + hW) * wa   (dominant pass: bf16 VW from L2)
    attn_score<<<(Bsz * Lann * 32 + 255) / 256, 256, 0, stream>>>(VWbf, hW, wa, escore);
    softmax_rows<<<Bsz, 256, 0, stream>>>(escore, alpha);
    // Ct -> concat.Ct
    ct_kernel<<<(Bsz * VL + 255) / 256, 256, 0, stream>>>(alpha, Vbf, CC);
    // O_t = tanh(W_O @ concat^T)  ->  Z.O (bf16, transposed store)
    wmma_gemm<2><<<gemm_grid(OSZ, Bsz), 128, 0, stream>>>(
        WOb, CLD, CC, CLD, (void*)(Zt + EMB), ZLD, nullptr, OSZ, Bsz, CLD);
    // logits = W_out @ O^T  ->  out[b][t][:]  (fp32, transposed store, ldd = SEQ*VOCAB)
    wmma_gemm<0><<<gemm_grid(VOCAB, Bsz), 128, 0, stream>>>(
        Woutb, OSZ, Zt + EMB, ZLD, (void*)(out + (size_t)t * VOCAB), SEQ * VOCAB,
        nullptr, VOCAB, Bsz, OSZ);
    // greedy feedback -> Z.emb (faithful row-set bug)
    next_token<<<1, 1024, 0, stream>>>(out + (size_t)t * VOCAB, E, Zt);
  }
}
